// CrossAttention_44384192036993
// MI455X (gfx1250) — matrix operand order repaired and checked by reading, not compiled, not run on hardware
//
#include <hip/hip_runtime.h>

// ---------------------------------------------------------------------------
// CrossAttention for MI455X (gfx1250, wave32, WMMA bf16 + TDM/async-LDS path)
// B=4, NQ=NK=2048, D=512, H=8, DH=64
//   1. convert inputs to bf16; transpose-convert weights to [N,K] bf16
//   2. Q = Xq@Wq, K = E@Wk (bf16 out), Vt = (E@Wv)^T
//      (WMMA GEMMs, tiles staged by the Tensor Data Mover, double buffered)
//   3. pass1: per-key column max / sum-exp over all queries (WMMA + shfl)
//   4. pass2: recompute S, p=exp(s-m)/l, rowsum, O = P@V
//      (K/V tiles async-DMA'd to LDS, double buffered, shared by 8 waves)
//   5. Y = bf16(Xq - (O@W0 + b0))        (WMMA GEMM, fused residual)
//   6. out = Y@W1 + b1  -> f32 d_out     (WMMA GEMM)
// ---------------------------------------------------------------------------

#define DEV static __device__ __forceinline__

typedef __bf16 bf16_t;
typedef __attribute__((ext_vector_type(16))) __bf16 v16bf;
typedef __attribute__((ext_vector_type(8)))  __bf16 v8bf;
typedef __attribute__((ext_vector_type(8)))  float  v8f;

typedef unsigned int       u32;
typedef unsigned long long u64;
typedef __attribute__((ext_vector_type(4))) u32 u32x4;
typedef __attribute__((ext_vector_type(8))) int i32x8;
typedef __attribute__((ext_vector_type(4))) int i32x4;

constexpr int cB  = 4;
constexpr int cNQ = 2048;
constexpr int cNK = 2048;
constexpr int cD  = 512;
constexpr int cH  = 8;
constexpr int cDH = 64;
constexpr int cMQ = cB * cNQ;   // 8192 flattened query rows
constexpr int cMK = cB * cNK;   // 8192 flattened key rows
constexpr int cHD = cH * cDH;   // 512 inner dim

#if defined(__has_builtin)
#if __has_builtin(__builtin_amdgcn_tensor_load_to_lds)
#define HAVE_TDM 1
#endif
#endif
#ifndef HAVE_TDM
#define HAVE_TDM 0
#endif

union FragU { v16bf v; v8bf h[2]; };

DEV v8f wmma_bf16(v16bf a, v16bf b, v8f c) {
  return __builtin_amdgcn_wmma_f32_16x16x32_bf16(false, a, false, b, (short)0, c,
                                                 false, false);
}

// ---- per-lane async DMA: one 16-byte chunk global -> LDS (ASYNCcnt) -------
// Generic pointers to __shared__ carry the wave-relative LDS offset in their
// low 32 bits (ISA 10.2 aperture mapping), which is exactly what VDST wants.
DEV void async_copy_b128(bf16_t* ldst, const bf16_t* gsrc) {
  unsigned laddr = (unsigned)(u64)ldst;
  asm volatile("global_load_async_to_lds_b128 %0, %1, off"
               :: "v"(laddr), "v"(gsrc) : "memory");
}
DEV void wait_asynccnt0() { asm volatile("s_wait_asynccnt 0" ::: "memory"); }
DEV void wait_dscnt0()    { asm volatile("s_wait_dscnt 0"    ::: "memory"); }

DEV void wait_tensorcnt0() {
#if defined(__has_builtin)
#if __has_builtin(__builtin_amdgcn_s_wait_tensorcnt)
  __builtin_amdgcn_s_wait_tensorcnt(0);
  asm volatile("" ::: "memory");
  return;
#endif
#endif
  asm volatile("s_wait_tensorcnt 0" ::: "memory");
}

#if HAVE_TDM
// ---- Tensor Data Mover: DMA a 2D bf16 tile (tileX x tileY elems) from a
// row-major tensor (strideElems between rows) into LDS at ldsAddr, inserting
// `padDw` DWORDs of padding after every `1<<(padIntervalCode+1)` DWORDs so
// the LDS layout matches our padded tiles.  D# per ISA 8.3/8.4.
DEV void tdm_load_2d(u32 ldsAddr, const bf16_t* gsrc,
                     u32 tileX, u32 tileY, u32 strideElems,
                     u32 padIntervalCode, u32 padDw) {
  const u64 ga = (u64)gsrc;
  u32x4 g0;
  g0[0] = 1u;                                   // count=1, user descriptor
  g0[1] = ldsAddr;                              // lds_addr
  g0[2] = (u32)ga;                              // global_addr[31:0]
  g0[3] = (u32)((ga >> 32) & 0x01FFFFFFu) | (2u << 30);  // addr[56:32], type=2
  i32x8 g1;
  u32 w0 = (1u << 16);                          // data_size = 2 bytes
  if (padDw) w0 |= (1u << 20) | (padIntervalCode << 22) | ((padDw - 1u) << 25);
  g1[0] = (int)w0;
  g1[1] = (int)((tileX & 0xFFFFu) << 16);       // tensor_dim0[15:0]
  g1[2] = (int)(((tileX >> 16) & 0xFFFFu) | ((tileY & 0xFFFFu) << 16));
  g1[3] = (int)(((tileY >> 16) & 0xFFFFu) | ((tileX & 0xFFFFu) << 16)); // tile_dim0
  g1[4] = (int)(tileY & 0xFFFFu);               // tile_dim1; tile_dim2=0
  g1[5] = (int)strideElems;                     // tensor_dim0_stride[31:0]
  g1[6] = 0;                                    // stride0 hi, stride1 lo
  g1[7] = 0;
  const i32x4 z4 = {0, 0, 0, 0};
#if __clang_major__ >= 23
  const i32x8 z8 = {0, 0, 0, 0, 0, 0, 0, 0};
  __builtin_amdgcn_tensor_load_to_lds(g0, g1, z4, z4, z8, 0);
#else
  __builtin_amdgcn_tensor_load_to_lds(g0, g1, z4, z4, 0);
#endif
}
#endif

// A-operand fragment: 16 rows x 32 K, bf16, row-major source.
DEV v16bf frag_load_A(const bf16_t* tile, int ld) {
  const int lane = threadIdx.x & 31;
  const int m = lane & 15, half = lane >> 4;
  const bf16_t* p = tile + (size_t)m * ld + half * 8;
  FragU f;
  f.h[0] = *(const v8bf*)(p);
  f.h[1] = *(const v8bf*)(p + 16);
  return f.v;
}

// B-operand fragment: 32 K x 16 cols, from N-major ([col][k] row-major) source.
DEV v16bf frag_load_B(const bf16_t* tile, int ld) {
  const int lane = threadIdx.x & 31;
  const int n = lane & 15, half = lane >> 4;
  const bf16_t* p = tile + (size_t)n * ld + half * 8;
  FragU f;
  f.h[0] = *(const v8bf*)(p);
  f.h[1] = *(const v8bf*)(p + 16);
  return f.v;
}

// ---------------------------------------------------------------------------
// conversion kernels
// ---------------------------------------------------------------------------
__global__ __launch_bounds__(256)
void cvt_f32_bf16_kernel(const float* __restrict__ src, bf16_t* __restrict__ dst, int n) {
  int i = blockIdx.x * blockDim.x + threadIdx.x;
  if (i < n) dst[i] = (bf16_t)src[i];
}

// src is [K,N] f32 row-major -> dst [N,K] bf16 row-major
__global__ __launch_bounds__(256)
void transpose_cvt_kernel(const float* __restrict__ src, bf16_t* __restrict__ dst,
                          int K, int N) {
  int i = blockIdx.x * blockDim.x + threadIdx.x;
  if (i >= K * N) return;
  int k = i / N, n = i % N;
  dst[(size_t)n * K + k] = (bf16_t)src[(size_t)k * N + n];
}

// ---------------------------------------------------------------------------
// bf16 WMMA GEMM: C[M,N] = A[M,K] @ Bt[N,K]^T  (+ epilogue)
// block 128 (4 waves), WG tile 64x64, wave tile 32x32, k-step 32.
// A/B 64x32 tiles double-buffered in LDS, filled by the Tensor Data Mover
// (one descriptor per tile, hardware pad to the 40-elem LDS row stride).
// EPI: 0 = store bf16          1 = store bf16 transposed (out[n*ldo+m])
//      2 = bf16( X - (acc+b) ) 3 = f32( acc + b )
// ---------------------------------------------------------------------------
template <int EPI>
__global__ __launch_bounds__(128)
void gemm_bf16_kernel(const bf16_t* __restrict__ A, int lda,
                      const bf16_t* __restrict__ Bt, int ldb,
                      int M, int N, int K,
                      const float* __restrict__ bias,
                      const float* __restrict__ X, int ldx,
                      bf16_t* __restrict__ outBf, float* __restrict__ outF, int ldo) {
  constexpr int LDT = 40;                    // padded LDS row stride (80 B)
  __shared__ bf16_t As[2][64][LDT];
  __shared__ bf16_t Bs[2][64][LDT];

  const int tid  = threadIdx.x;
  const int wave = tid >> 5, lane = tid & 31;
  const int l15  = lane & 15, half = lane >> 4;
  const int m0 = blockIdx.y * 64;
  const int n0 = blockIdx.x * 64;
  const int wm = (wave >> 1) * 32, wn = (wave & 1) * 32;

#if HAVE_TDM
  // one TDM descriptor per 64x32 tile: row = 32 elems = 16 DWORDs (code 3),
  // then 4 DWORDs pad -> LDS rows of 40 elems.
  auto stage = [&](int buf, int k0) {
    if (wave == 0) {
      tdm_load_2d((u32)(u64)&As[buf][0][0], A + (size_t)m0 * lda + k0,
                  32, 64, (u32)lda, 3, 4);
      tdm_load_2d((u32)(u64)&Bs[buf][0][0], Bt + (size_t)n0 * ldb + k0,
                  32, 64, (u32)ldb, 3, 4);
    }
  };
#else
  auto stage = [&](int buf, int k0) {
#pragma unroll
    for (int i = 0; i < 2; ++i) {
      const int c = tid * 2 + i;
      const int row = c >> 2, sub = c & 3;
      async_copy_b128(&As[buf][row][sub * 8],
                      A + (size_t)(m0 + row) * lda + k0 + sub * 8);
      async_copy_b128(&Bs[buf][row][sub * 8],
                      Bt + (size_t)(n0 + row) * ldb + k0 + sub * 8);
    }
  };
#endif

  stage(0, 0);
  v8f c[2][2] = {};
  const int iters = K / 32;
  for (int it = 0; it < iters; ++it) {
#if HAVE_TDM
    wait_tensorcnt0();        // wave0's DMA into buf (it&1) landed
#else
    wait_asynccnt0();
#endif
    __syncthreads();          // DMA visible to all waves; prev buffer fully read
    if (it + 1 < iters) stage((it + 1) & 1, (it + 1) * 32);
    const int buf = it & 1;
    v16bf a0 = frag_load_A(&As[buf][wm][0],      LDT);
    v16bf a1 = frag_load_A(&As[buf][wm + 16][0], LDT);
    v16bf b0 = frag_load_B(&Bs[buf][wn][0],      LDT);
    v16bf b1 = frag_load_B(&Bs[buf][wn + 16][0], LDT);
    c[0][0] = wmma_bf16(a0, b0, c[0][0]);
    c[0][1] = wmma_bf16(a0, b1, c[0][1]);
    c[1][0] = wmma_bf16(a1, b0, c[1][0]);
    c[1][1] = wmma_bf16(a1, b1, c[1][1]);
  }

#pragma unroll
  for (int i = 0; i < 2; ++i)
#pragma unroll
    for (int j = 0; j < 2; ++j)
#pragma unroll
      for (int r = 0; r < 8; ++r) {
        const int row = m0 + wm + i * 16 + r + half * 8;
        const int col = n0 + wn + j * 16 + l15;
        const float acc = c[i][j][r];
        if (EPI == 0) {
          outBf[(size_t)row * ldo + col] = (bf16_t)acc;
        } else if (EPI == 1) {
          outBf[(size_t)col * ldo + row] = (bf16_t)acc;
        } else if (EPI == 2) {
          float v = X[(size_t)row * ldx + col] - (acc + bias[col]);
          outBf[(size_t)row * ldo + col] = (bf16_t)v;
        } else {
          outF[(size_t)row * ldo + col] = acc + bias[col];
        }
      }
}

// ---------------------------------------------------------------------------
// pass 1: per-key (column) softmax stats over the query axis.
// One wave owns a 16-key tile of one (b,h); streams all queries with WMMA.
// Q/K are L2-resident (256 KB per head), so direct global fragment loads.
// ---------------------------------------------------------------------------
__global__ __launch_bounds__(256)
void attn_colstats_kernel(const bf16_t* __restrict__ Qb,  // [MQ, 512]
                          const bf16_t* __restrict__ Kb,  // [MK, 512]
                          float* __restrict__ colmax,     // [B*H, NK]
                          float* __restrict__ colsum) {
  const int lane = threadIdx.x & 31;
  const int l15 = lane & 15;
  const int wave = blockIdx.x * (blockDim.x >> 5) + (threadIdx.x >> 5);
  const int KT = cNK / 16;                 // 128 key tiles per head
  const int kt = wave % KT;
  const int bh = wave / KT;                // b*H + h
  const int h = bh % cH, b = bh / cH;

  const bf16_t* Qh = Qb + (size_t)b * cNQ * cHD + h * cDH;
  const bf16_t* Kh = Kb + (size_t)b * cNK * cHD + h * cDH;
  const int key0 = kt * 16;

  // key fragments are loop-invariant (B operand of S = Q @ K^T)
  const v16bf kb0 = frag_load_B(Kh + (size_t)key0 * cHD + 0,  cHD);
  const v16bf kb1 = frag_load_B(Kh + (size_t)key0 * cHD + 32, cHD);

  float m = -3.0e38f, l = 0.0f;
  for (int q0 = 0; q0 < cNQ; q0 += 16) {
    __builtin_prefetch(Qh + (size_t)(q0 + 16) * cHD, 0, 0);
    v16bf qa0 = frag_load_A(Qh + (size_t)q0 * cHD + 0,  cHD);
    v16bf qa1 = frag_load_A(Qh + (size_t)q0 * cHD + 32, cHD);
    v8f s = {};
    s = wmma_bf16(qa0, kb0, s);
    s = wmma_bf16(qa1, kb1, s);
    float tmax = s[0];
#pragma unroll
    for (int r = 1; r < 8; ++r) tmax = fmaxf(tmax, s[r]);
    const float mn = fmaxf(m, tmax);
    float acc = 0.0f;
#pragma unroll
    for (int r = 0; r < 8; ++r) acc += __expf(s[r] - mn);
    l = l * __expf(m - mn) + acc;
    m = mn;
  }
  // combine the two row-halves (lane c holds rows 0-7, lane c+16 rows 8-15)
  const float mo = __shfl_xor(m, 16, 32);
  const float lo = __shfl_xor(l, 16, 32);
  const float mn = fmaxf(m, mo);
  const float lt = l * __expf(m - mn) + lo * __expf(mo - mn);
  if (lane < 16) {
    colmax[(size_t)bh * cNK + key0 + l15] = mn;
    colsum[(size_t)bh * cNK + key0 + l15] = lt;
  }
}

// ---------------------------------------------------------------------------
// pass 2: recompute S, p = exp(s - colmax)/colsum, rowsum, O = P @ V.
// Block = 8 waves, all on the SAME (b,h), consecutive 16-query tiles.
// K (32x64) and V (64x32) tiles are async-DMA'd into double-buffered LDS and
// shared by all 8 waves (8x traffic reduction). P goes through a per-wave
// LDS tile to be re-read in A-operand layout.
// ---------------------------------------------------------------------------
__global__ __launch_bounds__(256)
void attn_out_kernel(const bf16_t* __restrict__ Qb,   // [MQ, 512]
                     const bf16_t* __restrict__ Kb,   // [MK, 512]
                     const bf16_t* __restrict__ Vt,   // [512, MK] (h*64+dh major)
                     const float* __restrict__ colmax,
                     const float* __restrict__ colsum,
                     bf16_t* __restrict__ attnOut) {  // [MQ, 512]
  constexpr int KLD = 72;   // 32 keys x 64 dh, padded  (144 B rows)
  constexpr int VLD = 40;   // 64 dh x 32 keys, padded  (80 B rows)
  constexpr int PLD = 40;   // per-wave P tile stride
  __shared__ bf16_t Ks[2][32][KLD];
  __shared__ bf16_t Vs[2][64][VLD];
  __shared__ bf16_t Ps[8][16 * PLD];

  const int tid = threadIdx.x;
  const int lane = tid & 31;
  const int l15 = lane & 15, half = lane >> 4;
  const int wslot = tid >> 5;
  const int wave = blockIdx.x * 8 + wslot;
  const int QT = cNQ / 16;                 // 128 query tiles per head
  const int qt = wave % QT;
  const int bh = wave / QT;                // all 8 waves of a block share bh
  const int h = bh % cH, b = bh / cH;

  const bf16_t* Qh = Qb + (size_t)b * cNQ * cHD + h * cDH;
  const bf16_t* Kh = Kb + (size_t)b * cNK * cHD + h * cDH;
  const bf16_t* Vh = Vt + (size_t)(h * cDH) * cMK + (size_t)b * cNK;
  const float* cmaxp = colmax + (size_t)bh * cNK;
  const float* csump = colsum + (size_t)bh * cNK;
  const int q0 = qt * 16;

  // stage K tile (32 rows x 64 elems = 256 chunks) and V tile (64 x 32 = 256)
  auto stage = [&](int buf, int n0) {
    {
      const int row = tid >> 3, sub = tid & 7;     // K: 8 chunks per row
      async_copy_b128(&Ks[buf][row][sub * 8],
                      Kh + (size_t)(n0 + row) * cHD + sub * 8);
    }
    {
      const int row = tid >> 2, sub = tid & 3;     // V: 4 chunks per row
      async_copy_b128(&Vs[buf][row][sub * 8],
                      Vh + (size_t)row * cMK + n0 + sub * 8);
    }
  };

  const v16bf qa0 = frag_load_A(Qh + (size_t)q0 * cHD + 0,  cHD);
  const v16bf qa1 = frag_load_A(Qh + (size_t)q0 * cHD + 32, cHD);

  v8f o[4] = {};
  float rs[8] = {};
  bf16_t* pt = &Ps[wslot][0];

  stage(0, 0);
  for (int n0 = 0; n0 < cNK; n0 += 32) {
    wait_asynccnt0();
    __syncthreads();
    if (n0 + 32 < cNK) stage(((n0 >> 5) + 1) & 1, n0 + 32);
    const int buf = (n0 >> 5) & 1;

#pragma unroll
    for (int sub = 0; sub < 2; ++sub) {
      const int key0 = n0 + sub * 16;
      v16bf kb0 = frag_load_B(&Ks[buf][sub * 16][0] + 0,  KLD);
      v16bf kb1 = frag_load_B(&Ks[buf][sub * 16][0] + 32, KLD);
      v8f s = {};
      s = wmma_bf16(qa0, kb0, s);
      s = wmma_bf16(qa1, kb1, s);
      const float mc = cmaxp[key0 + l15];
      const float inv = 1.0f / csump[key0 + l15];
#pragma unroll
      for (int r = 0; r < 8; ++r) {
        float p = __expf(s[r] - mc) * inv;   // column-softmax prob
        rs[r] += p;                          // row-sum partial (this lane's cols)
        pt[(r + half * 8) * PLD + sub * 16 + l15] = (bf16_t)p;
      }
    }
    wait_dscnt0();                           // P tile visible wave-wide
    const v16bf pa = frag_load_A(pt, PLD);
#pragma unroll
    for (int j = 0; j < 4; ++j) {
      v16bf vb = frag_load_B(&Vs[buf][j * 16][0], VLD);
      o[j] = wmma_bf16(pa, vb, o[j]);
    }
  }

  // reduce row sums across the 16 lanes of each half (rows r / 8+r)
#pragma unroll
  for (int r = 0; r < 8; ++r) {
    rs[r] += __shfl_xor(rs[r], 1, 32);
    rs[r] += __shfl_xor(rs[r], 2, 32);
    rs[r] += __shfl_xor(rs[r], 4, 32);
    rs[r] += __shfl_xor(rs[r], 8, 32);
  }

#pragma unroll
  for (int j = 0; j < 4; ++j)
#pragma unroll
    for (int r = 0; r < 8; ++r) {
      const float val = o[j][r] / (1e-12f + rs[r]);
      const size_t row = (size_t)b * cNQ + q0 + r + half * 8;
      attnOut[row * cHD + h * cDH + j * 16 + l15] = (bf16_t)val;
    }
}

// ---------------------------------------------------------------------------
// host launcher
// ---------------------------------------------------------------------------
extern "C" void kernel_launch(void* const* d_in, const int* in_sizes, int n_in,
                              void* d_out, int out_size, void* d_ws, size_t ws_size,
                              hipStream_t stream) {
  const float* init_query = (const float*)d_in[0];  // [B,NQ,D]
  const float* embedding  = (const float*)d_in[1];  // [B,NK,D]
  const float* Wq = (const float*)d_in[2];          // [D,HD]
  const float* Wk = (const float*)d_in[3];
  const float* Wv = (const float*)d_in[4];
  const float* W0 = (const float*)d_in[5];          // [HD,D]
  const float* b0 = (const float*)d_in[6];
  const float* W1 = (const float*)d_in[7];          // [D,D]
  const float* b1 = (const float*)d_in[8];
  float* out = (float*)d_out;

  uint8_t* ws = (uint8_t*)d_ws;
  size_t off = 0;
  auto alloc = [&](size_t bytes) -> void* {
    void* p = ws + off;
    off = (off + bytes + 255) & ~(size_t)255;
    return p;
  };

  const size_t actElems = (size_t)cMQ * cD;         // 8192*512
  const size_t wElems   = (size_t)cD * cD;          // 512*512
  bf16_t* Xq_bf = (bf16_t*)alloc(actElems * 2);
  bf16_t* E_bf  = (bf16_t*)alloc(actElems * 2);
  bf16_t* Wq_t  = (bf16_t*)alloc(wElems * 2);
  bf16_t* Wk_t  = (bf16_t*)alloc(wElems * 2);
  bf16_t* Wv_t  = (bf16_t*)alloc(wElems * 2);
  bf16_t* W0_t  = (bf16_t*)alloc(wElems * 2);
  bf16_t* W1_t  = (bf16_t*)alloc(wElems * 2);
  bf16_t* Qbf   = (bf16_t*)alloc(actElems * 2);
  bf16_t* Kbf   = (bf16_t*)alloc(actElems * 2);
  bf16_t* Vtb   = (bf16_t*)alloc(actElems * 2);     // [512, 8192]
  float*  cmax  = (float*)alloc((size_t)cB * cH * cNK * 4);
  float*  csum  = (float*)alloc((size_t)cB * cH * cNK * 4);
  bf16_t* attnO = (bf16_t*)alloc(actElems * 2);
  bf16_t* Ybf   = (bf16_t*)alloc(actElems * 2);
  (void)ws_size; (void)in_sizes; (void)n_in; (void)out_size;

  // 1. conversions
  {
    const int n = (int)actElems;
    cvt_f32_bf16_kernel<<<(n + 255) / 256, 256, 0, stream>>>(init_query, Xq_bf, n);
    cvt_f32_bf16_kernel<<<(n + 255) / 256, 256, 0, stream>>>(embedding, E_bf, n);
    const int nw = (int)wElems;
    transpose_cvt_kernel<<<(nw + 255) / 256, 256, 0, stream>>>(Wq, Wq_t, cD, cHD);
    transpose_cvt_kernel<<<(nw + 255) / 256, 256, 0, stream>>>(Wk, Wk_t, cD, cHD);
    transpose_cvt_kernel<<<(nw + 255) / 256, 256, 0, stream>>>(Wv, Wv_t, cD, cHD);
    transpose_cvt_kernel<<<(nw + 255) / 256, 256, 0, stream>>>(W0, W0_t, cHD, cD);
    transpose_cvt_kernel<<<(nw + 255) / 256, 256, 0, stream>>>(W1, W1_t, cD, cD);
  }

  // 2. projections
  {
    dim3 grid(cHD / 64, cMQ / 64);  // (8, 128)
    gemm_bf16_kernel<0><<<grid, 128, 0, stream>>>(Xq_bf, cD, Wq_t, cD,
        cMQ, cHD, cD, nullptr, nullptr, 0, Qbf, nullptr, cHD);
    gemm_bf16_kernel<0><<<grid, 128, 0, stream>>>(E_bf, cD, Wk_t, cD,
        cMK, cHD, cD, nullptr, nullptr, 0, Kbf, nullptr, cHD);
    gemm_bf16_kernel<1><<<grid, 128, 0, stream>>>(E_bf, cD, Wv_t, cD,
        cMK, cHD, cD, nullptr, nullptr, 0, Vtb, nullptr, cMK);
  }

  // 3. column softmax stats (over query axis)
  {
    const int waves = cB * cH * (cNK / 16);        // 4096
    attn_colstats_kernel<<<waves / 8, 256, 0, stream>>>(Qbf, Kbf, cmax, csum);
  }

  // 4. attention output
  {
    const int waves = cB * cH * (cNQ / 16);        // 4096
    attn_out_kernel<<<waves / 8, 256, 0, stream>>>(Qbf, Kbf, Vtb, cmax, csum, attnO);
  }

  // 5. Y = bf16( Xq - (attnO @ W0 + b0) )
  {
    dim3 grid(cD / 64, cMQ / 64);
    gemm_bf16_kernel<2><<<grid, 128, 0, stream>>>(attnO, cHD, W0_t, cHD,
        cMQ, cD, cHD, b0, init_query, cD, Ybf, nullptr, cD);
  }

  // 6. out = Y @ W1 + b1  (f32)
  {
    dim3 grid(cD / 64, cMQ / 64);
    gemm_bf16_kernel<3><<<grid, 128, 0, stream>>>(Ybf, cD, W1_t, cD,
        cMQ, cD, cD, b1, nullptr, 0, nullptr, out, cD);
  }
}
